// Prediction_Model_v4_GRU_4561255269212
// MI455X (gfx1250) — matrix operand order repaired and checked by reading, not compiled, not run on hardware
//
#include <hip/hip_runtime.h>
#include <hip/hip_bf16.h>

typedef __bf16 bf16_t;
typedef __attribute__((ext_vector_type(16))) __bf16 v16bf;
typedef __attribute__((ext_vector_type(8)))  __bf16 v8bf;
typedef __attribute__((ext_vector_type(8)))  float  v8f;

union ABFrag { v16bf v; v8bf h[2]; };

// ---------------------------------------------------------------------------
// WMMA fragment loaders (layouts per CDNA5 ISA 7.12.2, wave32)
// ---------------------------------------------------------------------------

// A-matrix 16x32 bf16, row-major source with leading dim `ld` (elements).
// lanes 0-15: m=lane, K 0..7 (VGPR0-3) and 16..23 (VGPR4-7)
// lanes16-31: m=lane-16, K 8..15 and 24..31
__device__ __forceinline__ v16bf load_a_bf16(const bf16_t* tile, int ld) {
    const int lane  = threadIdx.x & 31;
    const int m     = lane & 15;
    const int khalf = (lane >> 4) * 8;
    const bf16_t* p = tile + (size_t)m * ld + khalf;
    ABFrag u;
    u.h[0] = *(const v8bf*)(p);        // K = khalf .. khalf+7
    u.h[1] = *(const v8bf*)(p + 16);   // K = khalf+16 .. khalf+23
    return u.v;
}

// Same A layout but converting from fp32 source on the fly.
__device__ __forceinline__ v16bf load_a_f32(const float* tile, int ld) {
    const int lane  = threadIdx.x & 31;
    const int m     = lane & 15;
    const int khalf = (lane >> 4) * 8;
    const float* p = tile + (size_t)m * ld + khalf;
    ABFrag u;
#pragma unroll
    for (int i = 0; i < 8; ++i) {
        u.h[0][i] = (bf16_t)p[i];
        u.h[1][i] = (bf16_t)p[16 + i];
    }
    return u.v;
}

// B-matrix 32x16 bf16 for C = A * W^T with W row-major [N, K]:
// lane n = lane&15; lanes 0-15 hold K k0..k0+7 and k0+16..k0+23,
// lanes16-31 hold k0+8..k0+15 and k0+24..k0+31 (same K order as A).
__device__ __forceinline__ v16bf load_b_bf16(const bf16_t* W, int ldk, int n0, int k0) {
    const int lane = threadIdx.x & 31;
    const bf16_t* p = W + (size_t)(n0 + (lane & 15)) * ldk + k0 + ((lane >> 4) << 3);
    ABFrag u;
    u.h[0] = *(const v8bf*)(p);
    u.h[1] = *(const v8bf*)(p + 16);
    return u.v;
}

// Same B layout converting from fp32 weights on the fly (small FC matrices).
__device__ __forceinline__ v16bf load_b_f32(const float* W, int ldk, int n0, int k0) {
    const int lane = threadIdx.x & 31;
    const float* p = W + (size_t)(n0 + (lane & 15)) * ldk + k0 + ((lane >> 4) << 3);
    ABFrag u;
#pragma unroll
    for (int i = 0; i < 8; ++i) {
        u.h[0][i] = (bf16_t)p[i];
        u.h[1][i] = (bf16_t)p[16 + i];
    }
    return u.v;
}

__device__ __forceinline__ v8f wmma_bf16(v16bf a, v16bf b, v8f c) {
    // D = A(16x32 bf16) x B(32x16 bf16) + C(16x16 f32)
    return __builtin_amdgcn_wmma_f32_16x16x32_bf16(
        /*neg_a=*/false, a, /*neg_b=*/false, b,
        /*c_mod=*/(short)0, c, /*reuse_a=*/false, /*reuse_b=*/false);
}

// C/D 16x16 f32: lane n = lane&15, rows mbase+i (mbase = 8*(lane>=16)), VGPR i.
__device__ __forceinline__ void store_c_f32(float* C, int ldn, v8f acc) {
    const int lane  = threadIdx.x & 31;
    const int n     = lane & 15;
    const int mbase = (lane >> 4) * 8;
#pragma unroll
    for (int i = 0; i < 8; ++i) C[(size_t)(mbase + i) * ldn + n] = acc[i];
}

__device__ __forceinline__ float sigmoid_f(float x) {
    return 1.0f / (1.0f + __expf(-x));
}
__device__ __forceinline__ float tanh_f(float x) {
    return 2.0f / (1.0f + __expf(-2.0f * x)) - 1.0f;
}

// ---------------------------------------------------------------------------
// Kernel 1: convert the four [1536,512] fp32 weight matrices to bf16.
// ---------------------------------------------------------------------------
__global__ void k_cvt_weights(const float* __restrict__ s0, const float* __restrict__ s1,
                              const float* __restrict__ s2, const float* __restrict__ s3,
                              bf16_t* __restrict__ dst) {
    const int i = blockIdx.x * blockDim.x + threadIdx.x;
    const int N = 1536 * 512;
    if (i < N) {
        dst[i]         = (bf16_t)s0[i];
        dst[N + i]     = (bf16_t)s1[i];
        dst[2 * N + i] = (bf16_t)s2[i];
        dst[3 * N + i] = (bf16_t)s3[i];
    }
}

// ---------------------------------------------------------------------------
// Kernel 2: XG[M,1536] = A[M,512] @ W_ih^T + b_ih   (stored bf16)
//   grid.x = M/16 (row tiles), grid.y = 12, block = 256 (8 waves),
//   each wave owns one 16x16 output tile; K-loop of 16 bf16 WMMAs.
// ---------------------------------------------------------------------------
template <bool A_IS_F32>
__global__ __launch_bounds__(256) void k_xg_gemm(const void* __restrict__ A_,
                                                 const bf16_t* __restrict__ Wb,  // [1536,512] bf16
                                                 const float* __restrict__ bias, // [1536]
                                                 bf16_t* __restrict__ XG,        // [M,1536] bf16
                                                 int K, int N3) {
    const int wave = threadIdx.x >> 5;
    const int mt   = blockIdx.x;
    const int n0   = (blockIdx.y * 8 + wave) * 16;

    v8f acc = {};
#pragma unroll 4
    for (int k = 0; k < 512; k += 32) {
        v16bf a;
        if constexpr (A_IS_F32)
            a = load_a_f32((const float*)A_ + (size_t)mt * 16 * K + k, K);
        else
            a = load_a_bf16((const bf16_t*)A_ + (size_t)mt * 16 * K + k, K);
        v16bf b = load_b_bf16(Wb, K, n0, k);
        acc = wmma_bf16(a, b, acc);
    }
    const int lane  = threadIdx.x & 31;
    const int n     = lane & 15;
    const int mbase = (lane >> 4) * 8;
    const float bv  = bias[n0 + n];
#pragma unroll
    for (int i = 0; i < 8; ++i)
        XG[(size_t)(mt * 16 + mbase + i) * N3 + n0 + n] = (bf16_t)(acc[i] + bv);
}

// ---------------------------------------------------------------------------
// Kernel 3: GRU recurrence for one layer.
//   GRU is elementwise in batch -> one WG per 16-row batch tile (grid=8),
//   no cross-WG sync. 512 threads = 16 waves; per step each wave computes
//   6 N-tiles of hg = h @ W_hh^T (96 WMMAs/wave/step), hg staged in LDS,
//   then fused gate math updates h in LDS.
// ---------------------------------------------------------------------------
__global__ __launch_bounds__(512) void k_gru_layer(const bf16_t* __restrict__ XG,   // [B*T,1536]
                                                   const bf16_t* __restrict__ Whh,  // [1536,512] bf16
                                                   const float* __restrict__ b_hh,  // [1536]
                                                   bf16_t* __restrict__ h_seq,      // [B*T,512] bf16 or null
                                                   float* __restrict__ h_last,      // [B,512] f32 or null
                                                   int T) {
    __shared__ __align__(16) float  h_f[16 * 512];   // 32 KB  fp32 hidden state
    __shared__ __align__(16) bf16_t h_b[16 * 512];   // 16 KB  bf16 copy (WMMA A source)
    __shared__ __align__(16) float  hg[16 * 1536];   // 96 KB  h @ W_hh^T

    const int tid   = threadIdx.x;
    const int wave  = tid >> 5;
    const int btile = blockIdx.x;                    // batch rows btile*16 .. +15

    for (int i = tid; i < 16 * 512; i += 512) { h_f[i] = 0.0f; h_b[i] = (bf16_t)0.0f; }
    __syncthreads();

    for (int t = 0; t < T; ++t) {
        // prefetch this timestep's XG rows into cache while doing the GEMM
        if (tid < 16)
            __builtin_prefetch(XG + ((size_t)(btile * 16 + tid) * T + t) * 1536, 0, 3);

        // hg[16,1536] = h_b[16,512] @ Whh^T  -- 96 tiles over 16 waves
#pragma unroll
        for (int j = 0; j < 6; ++j) {
            const int n0 = (wave * 6 + j) * 16;
            v8f acc = {};
#pragma unroll 4
            for (int k = 0; k < 512; k += 32) {
                v16bf a = load_a_bf16(h_b + k, 512);       // from LDS
                v16bf b = load_b_bf16(Whh, 512, n0, k);    // from L2-resident weights
                acc = wmma_bf16(a, b, acc);
            }
            store_c_f32(hg + n0, 1536, acc);
        }
        __syncthreads();

        // fused gates: r,z,n + state update. 8192 elems / 512 threads.
        for (int i = tid; i < 16 * 512; i += 512) {
            const int m = i >> 9;
            const int c = i & 511;
            const size_t row = (size_t)(btile * 16 + m) * T + t;
            const bf16_t* xgp = XG + row * 1536;
            const float xr = (float)xgp[c];
            const float xz = (float)xgp[512 + c];
            const float xn = (float)xgp[1024 + c];
            const float hr = hg[m * 1536 + c];
            const float hz = hg[m * 1536 + 512 + c];
            const float hn = hg[m * 1536 + 1024 + c];
            const float r = sigmoid_f(xr + hr + b_hh[c]);
            const float z = sigmoid_f(xz + hz + b_hh[512 + c]);
            const float n = tanh_f(xn + r * (hn + b_hh[1024 + c]));
            const float hnew = (1.0f - z) * n + z * h_f[i];
            h_f[i] = hnew;
            h_b[i] = (bf16_t)hnew;
            if (h_seq) h_seq[row * 512 + c] = (bf16_t)hnew;
        }
        __syncthreads();
    }

    if (h_last) {
        for (int i = tid; i < 16 * 512; i += 512) {
            const int m = i >> 9;
            const int c = i & 511;
            h_last[(size_t)(btile * 16 + m) * 512 + c] = h_f[i];
        }
    }
}

// ---------------------------------------------------------------------------
// Kernel 4: fused head. out = h_last @ W_fc^T + b_fc ; oc = out @ W_fcc^T + b_fcc
//   Single WG (8 waves); intermediate `out` kept in LDS as bf16 for GEMM 2.
// ---------------------------------------------------------------------------
__global__ __launch_bounds__(256) void k_fc_head(const float* __restrict__ h,      // [128,512]
                                                 const float* __restrict__ Wfc,    // [128,512]
                                                 const float* __restrict__ bfc,    // [128]
                                                 const float* __restrict__ Wfcc,   // [384,128]
                                                 const float* __restrict__ bfcc,   // [384]
                                                 float* __restrict__ out,          // [128,128]
                                                 float* __restrict__ out_cate) {   // [128,384]
    __shared__ __align__(16) bf16_t ob[128 * 128];   // 32 KB

    const int wave  = threadIdx.x >> 5;
    const int lane  = threadIdx.x & 31;
    const int n     = lane & 15;
    const int mbase = (lane >> 4) * 8;
    const int mt    = wave;  // 0..7 -> row tile

    // GEMM 1: [128,512] x [512,128]
    for (int nt = 0; nt < 8; ++nt) {
        v8f acc = {};
#pragma unroll 4
        for (int k = 0; k < 512; k += 32) {
            v16bf a = load_a_f32(h + (size_t)(mt * 16) * 512 + k, 512);
            v16bf b = load_b_f32(Wfc, 512, nt * 16, k);
            acc = wmma_bf16(a, b, acc);
        }
        const float bv = bfc[nt * 16 + n];
#pragma unroll
        for (int i = 0; i < 8; ++i) {
            const float v = acc[i] + bv;
            const int m = mt * 16 + mbase + i;
            const int c = nt * 16 + n;
            out[m * 128 + c] = v;
            ob[m * 128 + c]  = (bf16_t)v;
        }
    }
    __syncthreads();

    // GEMM 2: [128,128] x [128,384]
    for (int nt = 0; nt < 24; ++nt) {
        v8f acc = {};
#pragma unroll
        for (int k = 0; k < 128; k += 32) {
            v16bf a = load_a_bf16(ob + (size_t)(mt * 16) * 128 + k, 128);
            v16bf b = load_b_f32(Wfcc, 128, nt * 16, k);
            acc = wmma_bf16(a, b, acc);
        }
        const float bv = bfcc[nt * 16 + n];
#pragma unroll
        for (int i = 0; i < 8; ++i)
            out_cate[(size_t)(mt * 16 + mbase + i) * 384 + nt * 16 + n] = acc[i] + bv;
    }
}

// ---------------------------------------------------------------------------
// Host-side orchestration
// ---------------------------------------------------------------------------
extern "C" void kernel_launch(void* const* d_in, const int* in_sizes, int n_in,
                              void* d_out, int out_size, void* d_ws, size_t ws_size,
                              hipStream_t stream) {
    constexpr int B = 128, T = 512, H = 512, O = 128;
    constexpr int M = B * T;            // 65536 GEMM rows
    constexpr int N3 = 3 * H;           // 1536

    const float* x     = (const float*)d_in[0];
    const float* W_ih0 = (const float*)d_in[1];
    const float* W_hh0 = (const float*)d_in[2];
    const float* b_ih0 = (const float*)d_in[3];
    const float* b_hh0 = (const float*)d_in[4];
    const float* W_ih1 = (const float*)d_in[5];
    const float* W_hh1 = (const float*)d_in[6];
    const float* b_ih1 = (const float*)d_in[7];
    const float* b_hh1 = (const float*)d_in[8];
    const float* W_fc  = (const float*)d_in[9];
    const float* b_fc  = (const float*)d_in[10];
    const float* W_fcc = (const float*)d_in[11];
    const float* b_fcc = (const float*)d_in[12];
    float* out = (float*)d_out;

    // workspace layout (bytes)
    char* ws = (char*)d_ws;
    const size_t XG_BYTES  = (size_t)M * N3 * sizeof(bf16_t);    // 201.3 MB (shared by both layers)
    const size_t H1_BYTES  = (size_t)M * H * sizeof(bf16_t);     //  67.1 MB
    const size_t WB_BYTES  = (size_t)4 * N3 * H * sizeof(bf16_t);//   6.3 MB
    bf16_t* XG    = (bf16_t*)ws;
    bf16_t* h1b   = (bf16_t*)(ws + XG_BYTES);
    bf16_t* Wb    = (bf16_t*)(ws + XG_BYTES + H1_BYTES);
    float*  h2l   = (float*) (ws + XG_BYTES + H1_BYTES + WB_BYTES);
    bf16_t* Wih0b = Wb;
    bf16_t* Whh0b = Wb + (size_t)N3 * H;
    bf16_t* Wih1b = Wb + (size_t)2 * N3 * H;
    bf16_t* Whh1b = Wb + (size_t)3 * N3 * H;

    // 1) weights -> bf16
    k_cvt_weights<<<dim3((N3 * H + 255) / 256), 256, 0, stream>>>(W_ih0, W_hh0, W_ih1, W_hh1, Wb);

    // 2) layer 0: XG = x @ W_ih0^T + b_ih0
    k_xg_gemm<true><<<dim3(M / 16, 12), 256, 0, stream>>>(x, Wih0b, b_ih0, XG, H, N3);

    // 3) layer 0 recurrence -> h1 (bf16 sequence)
    k_gru_layer<<<dim3(B / 16), 512, 0, stream>>>(XG, Whh0b, b_hh0, h1b, nullptr, T);

    // 4) layer 1: XG = h1 @ W_ih1^T + b_ih1 (reuse XG buffer)
    k_xg_gemm<false><<<dim3(M / 16, 12), 256, 0, stream>>>(h1b, Wih1b, b_ih1, XG, H, N3);

    // 5) layer 1 recurrence -> h2 last state only
    k_gru_layer<<<dim3(B / 16), 512, 0, stream>>>(XG, Whh1b, b_hh1, nullptr, h2l, T);

    // 6) fused FC head -> d_out[0:16384] = out, d_out[16384:65536] = out_cate
    k_fc_head<<<dim3(1), 256, 0, stream>>>(h2l, W_fc, b_fc, W_fcc, b_fcc, out, out + B * O);
}
